// TransformerEncoderLayer_68161130988098
// MI455X (gfx1250) — matrix operand order repaired and checked by reading, not compiled, not run on hardware
//
#include <hip/hip_runtime.h>
#include <math.h>

// Problem constants (match reference)
#define B_  2
#define S_  2048
#define D_  1024
#define H_  16
#define DK_ 64
#define FF_ 4096
#define NTOK (B_ * S_)   // 4096

typedef __attribute__((ext_vector_type(16))) _Float16 v16h;
typedef __attribute__((ext_vector_type(8)))  _Float16 v8h;
typedef __attribute__((ext_vector_type(8)))  float    v8f;

// ---------------------------------------------------------------------------
// WMMA helpers: v_wmma_f32_16x16x32_f16,  D = A(16x32 f16) * B(32x16 f16) + C
// Fragment layouts per CDNA5 ISA 7.12.2 (wave32):
//  A: lane<16 -> row M=lane, K = [k0..k0+7] and [k0+16..k0+23]
//     lane>=16 -> row M=lane-16, K = [k0+8..k0+15] and [k0+24..k0+31]
//  B: lane<16 -> col N=lane, K = [k0..k0+7] and [k0+16..k0+23]
//     lane>=16 -> col N=lane-16, K = [k0+8..k0+15] and [k0+24..k0+31]
//  C/D: vgpr i -> row M = i + (lane>=16 ? 8 : 0), col N = lane&15
// ---------------------------------------------------------------------------
static __device__ __forceinline__ v8f wmma_f16(v16h a, v16h b, v8f c) {
  return __builtin_amdgcn_wmma_f32_16x16x32_f16(false, a, false, b, (short)0, c,
                                                false, false);
}

static __device__ __forceinline__ v16h frag_a_f16(const _Float16* __restrict__ p,
                                                  int ld, int row0, int k0) {
  const int lane = threadIdx.x & 31;
  const _Float16* q = p + (size_t)(row0 + (lane & 15)) * ld + (k0 + ((lane >> 4) << 3));
  union { v16h v; v8h h[2]; } u;
  u.h[0] = *(const v8h*)(q);        // K +0..7
  u.h[1] = *(const v8h*)(q + 16);   // K +16..23
  return u.v;
}

static __device__ __forceinline__ v16h frag_b_f16(const _Float16* __restrict__ p,
                                                  int ld, int col0, int k0) {
  // p is "B-transposed": p[col*ld + k]  (contraction contiguous)
  const int lane = threadIdx.x & 31;
  const _Float16* q = p + (size_t)(col0 + (lane & 15)) * ld + (k0 + ((lane >> 4) << 3));
  union { v16h v; v8h h[2]; } u;
  u.h[0] = *(const v8h*)(q);        // K +0..7
  u.h[1] = *(const v8h*)(q + 16);   // K +16..23
  return u.v;
}

static __device__ __forceinline__ v16h frag_a_f32(const float* __restrict__ p,
                                                  int ld, int row0, int k0) {
  const int lane = threadIdx.x & 31;
  const float* q = p + (size_t)(row0 + (lane & 15)) * ld + (k0 + ((lane >> 4) << 3));
  union { v16h v; _Float16 e[16]; } u;
#pragma unroll
  for (int i = 0; i < 8; ++i) {
    u.e[i]     = (_Float16)q[i];       // K +0..7
    u.e[8 + i] = (_Float16)q[16 + i];  // K +16..23
  }
  return u.v;
}

// ---------------------------------------------------------------------------
// Weight convert + transpose: W fp32 [K,N] -> Wt f16 [N,K]
// ---------------------------------------------------------------------------
__global__ void k_cvtw(const float* __restrict__ W, _Float16* __restrict__ Wt,
                       int K, int N) {
  size_t t = (size_t)blockIdx.x * 256 + threadIdx.x;
  if (t >= (size_t)K * N) return;
  int k = (int)(t / N), n = (int)(t % N);
  Wt[(size_t)n * K + k] = (_Float16)W[t];
}

// ---------------------------------------------------------------------------
// LayerNorm over rows of length D_: fp32 in -> f16 out
// ---------------------------------------------------------------------------
__global__ void k_ln(const float* __restrict__ X, const float* __restrict__ g,
                     const float* __restrict__ b, _Float16* __restrict__ Y) {
  __shared__ float red[256];
  const int tid = threadIdx.x;
  const float* x = X + (size_t)blockIdx.x * D_;
  float ls = 0.f;
  for (int i = tid; i < D_; i += 256) ls += x[i];
  red[tid] = ls; __syncthreads();
  for (int s = 128; s > 0; s >>= 1) { if (tid < s) red[tid] += red[tid + s]; __syncthreads(); }
  const float mean = red[0] * (1.f / D_);
  __syncthreads();
  float lv = 0.f;
  for (int i = tid; i < D_; i += 256) { float d = x[i] - mean; lv += d * d; }
  red[tid] = lv; __syncthreads();
  for (int s = 128; s > 0; s >>= 1) { if (tid < s) red[tid] += red[tid + s]; __syncthreads(); }
  const float rstd = rsqrtf(red[0] * (1.f / D_) + 1e-6f);
  _Float16* y = Y + (size_t)blockIdx.x * D_;
  for (int i = tid; i < D_; i += 256)
    y[i] = (_Float16)((x[i] - mean) * rstd * g[i] + b[i]);
}

// ---------------------------------------------------------------------------
// Row softmax in place, row length S_ (2048), one block per row
// ---------------------------------------------------------------------------
__global__ void k_softmax(float* __restrict__ A) {
  __shared__ float red[256];
  const int tid = threadIdx.x;
  float* x = A + (size_t)blockIdx.x * S_;
  float v[S_ / 256];
  float lm = -3.0e38f;
#pragma unroll
  for (int i = 0; i < S_ / 256; ++i) { v[i] = x[tid + i * 256]; lm = fmaxf(lm, v[i]); }
  red[tid] = lm; __syncthreads();
  for (int s = 128; s > 0; s >>= 1) { if (tid < s) red[tid] = fmaxf(red[tid], red[tid + s]); __syncthreads(); }
  const float m = red[0];
  __syncthreads();
  float lsum = 0.f;
#pragma unroll
  for (int i = 0; i < S_ / 256; ++i) { v[i] = expf(v[i] - m); lsum += v[i]; }
  red[tid] = lsum; __syncthreads();
  for (int s = 128; s > 0; s >>= 1) { if (tid < s) red[tid] += red[tid + s]; __syncthreads(); }
  const float inv = 1.f / red[0];
#pragma unroll
  for (int i = 0; i < S_ / 256; ++i) x[tid + i * 256] = v[i] * inv;
}

// ---------------------------------------------------------------------------
// Double-buffered WMMA K-loop body (f16 A).  acc[0..3] over a 16x64 tile.
// Loads for step k+32 are issued before the WMMAs of step k.
// ---------------------------------------------------------------------------
#define GEMM_K_LOOP_F16(A_, lda_, Bt_, ldb_, row0_, col0_, K_)                 \
  v16h a  = frag_a_f16(A_, lda_, row0_, 0);                                    \
  v16h b0 = frag_b_f16(Bt_, ldb_, (col0_),      0);                            \
  v16h b1 = frag_b_f16(Bt_, ldb_, (col0_) + 16, 0);                            \
  v16h b2 = frag_b_f16(Bt_, ldb_, (col0_) + 32, 0);                            \
  v16h b3 = frag_b_f16(Bt_, ldb_, (col0_) + 48, 0);                            \
  for (int k0 = 0; k0 < (K_) - 32; k0 += 32) {                                 \
    const int kn = k0 + 32;                                                    \
    v16h an  = frag_a_f16(A_, lda_, row0_, kn);                                \
    v16h bn0 = frag_b_f16(Bt_, ldb_, (col0_),      kn);                        \
    v16h bn1 = frag_b_f16(Bt_, ldb_, (col0_) + 16, kn);                        \
    v16h bn2 = frag_b_f16(Bt_, ldb_, (col0_) + 32, kn);                        \
    v16h bn3 = frag_b_f16(Bt_, ldb_, (col0_) + 48, kn);                        \
    acc[0] = wmma_f16(a, b0, acc[0]);                                          \
    acc[1] = wmma_f16(a, b1, acc[1]);                                          \
    acc[2] = wmma_f16(a, b2, acc[2]);                                          \
    acc[3] = wmma_f16(a, b3, acc[3]);                                          \
    a = an; b0 = bn0; b1 = bn1; b2 = bn2; b3 = bn3;                            \
  }                                                                            \
  acc[0] = wmma_f16(a, b0, acc[0]);                                            \
  acc[1] = wmma_f16(a, b1, acc[1]);                                            \
  acc[2] = wmma_f16(a, b2, acc[2]);                                            \
  acc[3] = wmma_f16(a, b3, acc[3]);

// ---------------------------------------------------------------------------
// GEMM: C_f16[M,N] = A_f16[M,K] * Bt_f16[N,K]^T + bias.  Wave -> 16x64 tile.
// ---------------------------------------------------------------------------
__global__ void k_gemm_f16(const _Float16* __restrict__ A, int lda,
                           const _Float16* __restrict__ Bt, int ldb,
                           const float* __restrict__ bias,
                           _Float16* __restrict__ C, int ldc,
                           int ntile64, int K) {
  const int wave = blockIdx.x * 8 + (threadIdx.x >> 5);
  const int row0 = (wave / ntile64) << 4;
  const int col0 = (wave % ntile64) << 6;
  v8f acc[4] = {};
  GEMM_K_LOOP_F16(A, lda, Bt, ldb, row0, col0, K)
  const int lane = threadIdx.x & 31;
  const int r = row0 + ((lane >> 4) << 3);
#pragma unroll
  for (int j = 0; j < 4; ++j) {
    const int c = col0 + j * 16 + (lane & 15);
    const float bv = bias[c];
#pragma unroll
    for (int i = 0; i < 8; ++i)
      C[(size_t)(r + i) * ldc + c] = (_Float16)(acc[j][i] + bv);
  }
}

// Same GEMM, but stores V transposed: vT[b, h, d, s]  (key-contiguous rows)
__global__ void k_gemm_vT(const _Float16* __restrict__ A,
                          const _Float16* __restrict__ Bt,
                          const float* __restrict__ bias,
                          _Float16* __restrict__ VT) {
  const int wave = blockIdx.x * 8 + (threadIdx.x >> 5);
  const int row0 = (wave >> 4) << 4;          // token tile (ntile64 = 16)
  const int col0 = (wave & 15) << 6;
  v8f acc[4] = {};
  GEMM_K_LOOP_F16(A, D_, Bt, D_, row0, col0, D_)
  const int lane = threadIdx.x & 31;
  const int r = row0 + ((lane >> 4) << 3);
#pragma unroll
  for (int j = 0; j < 4; ++j) {
    const int cg = col0 + j * 16 + (lane & 15);   // h*64 + d
    const int h = cg >> 6, d = cg & 63;
    const float bv = bias[cg];
#pragma unroll
    for (int i = 0; i < 8; ++i) {
      const int t = r + i;                        // global token
      const int b = t >> 11, s = t & (S_ - 1);
      VT[(((size_t)b * H_ + h) * DK_ + d) * S_ + s] = (_Float16)(acc[j][i] + bv);
    }
  }
}

// GEMM with fp32 residual-add epilogue: C_f32 = res + A_f16*Bt^T + bias
__global__ void k_gemm_resid(const _Float16* __restrict__ A, int lda,
                             const _Float16* __restrict__ Bt, int ldb,
                             const float* __restrict__ bias,
                             const float* __restrict__ res,
                             float* __restrict__ C, int ldc,
                             int ntile64, int K) {
  const int wave = blockIdx.x * 8 + (threadIdx.x >> 5);
  const int row0 = (wave / ntile64) << 4;
  const int col0 = (wave % ntile64) << 6;
  v8f acc[4] = {};
  GEMM_K_LOOP_F16(A, lda, Bt, ldb, row0, col0, K)
  const int lane = threadIdx.x & 31;
  const int r = row0 + ((lane >> 4) << 3);
#pragma unroll
  for (int j = 0; j < 4; ++j) {
    const int c = col0 + j * 16 + (lane & 15);
    const float bv = bias[c];
#pragma unroll
    for (int i = 0; i < 8; ++i) {
      const size_t idx = (size_t)(r + i) * ldc + c;
      C[idx] = res[idx] + acc[j][i] + bv;
    }
  }
}

// scores[b,h,q,k] = (Q . K^T) / 8   per (b,h) via blockIdx.y  (K=64: 2 steps)
__global__ void k_scores(const _Float16* __restrict__ Q16,
                         const _Float16* __restrict__ K16,
                         float* __restrict__ attn) {
  const int bh = blockIdx.y, b = bh >> 4, h = bh & 15;
  const int wave = blockIdx.x * 8 + (threadIdx.x >> 5);
  const int row0 = (wave >> 5) << 4;       // 128 q tiles
  const int col0 = (wave & 31) << 6;       // 32 key tiles of 64
  const _Float16* A  = Q16 + (size_t)b * S_ * D_ + (size_t)h * DK_;
  const _Float16* Bt = K16 + (size_t)b * S_ * D_ + (size_t)h * DK_;
  v8f acc[4] = {};
#pragma unroll
  for (int k0 = 0; k0 < DK_; k0 += 32) {
    v16h a = frag_a_f16(A, D_, row0, k0);
#pragma unroll
    for (int j = 0; j < 4; ++j)
      acc[j] = wmma_f16(a, frag_b_f16(Bt, D_, col0 + j * 16, k0), acc[j]);
  }
  float* O = attn + (size_t)bh * S_ * S_;
  const int lane = threadIdx.x & 31;
  const int r = row0 + ((lane >> 4) << 3);
#pragma unroll
  for (int j = 0; j < 4; ++j) {
    const int c = col0 + j * 16 + (lane & 15);
#pragma unroll
    for (int i = 0; i < 8; ++i)
      O[(size_t)(r + i) * S_ + c] = acc[j][i] * 0.125f;  // 1/sqrt(64)
  }
}

// ctx16[token, h*64+d] = attn[b,h,q,:] . V  using vT[b,h,d,s]; K = S_ = 2048
__global__ void k_ctx(const float* __restrict__ attn,
                      const _Float16* __restrict__ VT,
                      _Float16* __restrict__ CTX) {
  const int bh = blockIdx.y, b = bh >> 4, h = bh & 15;
  const int wave = blockIdx.x * 8 + (threadIdx.x >> 5);
  const int row0 = wave << 4;                       // 128 q tiles, N = 64 = 4 tiles
  const float* A = attn + (size_t)bh * S_ * S_;
  const _Float16* Bt = VT + (size_t)bh * DK_ * S_;  // rows: d, contiguous over keys
  v8f acc[4] = {};
  v16h a  = frag_a_f32(A, S_, row0, 0);
  v16h b0 = frag_b_f16(Bt, S_, 0,  0);
  v16h b1 = frag_b_f16(Bt, S_, 16, 0);
  v16h b2 = frag_b_f16(Bt, S_, 32, 0);
  v16h b3 = frag_b_f16(Bt, S_, 48, 0);
  for (int k0 = 0; k0 < S_ - 32; k0 += 32) {
    const int kn = k0 + 32;
    v16h an  = frag_a_f32(A, S_, row0, kn);
    v16h bn0 = frag_b_f16(Bt, S_, 0,  kn);
    v16h bn1 = frag_b_f16(Bt, S_, 16, kn);
    v16h bn2 = frag_b_f16(Bt, S_, 32, kn);
    v16h bn3 = frag_b_f16(Bt, S_, 48, kn);
    acc[0] = wmma_f16(a, b0, acc[0]);
    acc[1] = wmma_f16(a, b1, acc[1]);
    acc[2] = wmma_f16(a, b2, acc[2]);
    acc[3] = wmma_f16(a, b3, acc[3]);
    a = an; b0 = bn0; b1 = bn1; b2 = bn2; b3 = bn3;
  }
  acc[0] = wmma_f16(a, b0, acc[0]);
  acc[1] = wmma_f16(a, b1, acc[1]);
  acc[2] = wmma_f16(a, b2, acc[2]);
  acc[3] = wmma_f16(a, b3, acc[3]);
  const int lane = threadIdx.x & 31;
  const int r = row0 + ((lane >> 4) << 3);
#pragma unroll
  for (int j = 0; j < 4; ++j) {
    const int c = h * 64 + j * 16 + (lane & 15);
#pragma unroll
    for (int i = 0; i < 8; ++i)
      CTX[(size_t)(b * S_ + r + i) * D_ + c] = (_Float16)acc[j][i];
  }
}

// Fused GeGLU: hh = gelu(A*wi0 + bi0) * (A*wi1 + bi1), f16 out [M, FF]
__global__ void k_geglu(const _Float16* __restrict__ A,
                        const _Float16* __restrict__ B0t,
                        const _Float16* __restrict__ B1t,
                        const float* __restrict__ bi0,
                        const float* __restrict__ bi1,
                        _Float16* __restrict__ HH) {
  const int wave = blockIdx.x * 8 + (threadIdx.x >> 5);
  const int row0 = (wave >> 6) << 4;     // FF/64 = 64 column tiles
  const int col0 = (wave & 63) << 6;
  v8f a0[4] = {}, a1[4] = {};
  v16h a = frag_a_f16(A, D_, row0, 0);
  for (int k0 = 0; k0 < D_; k0 += 32) {
    const int kn = k0 + 32;
    v16h an = (kn < D_) ? frag_a_f16(A, D_, row0, kn) : a;
#pragma unroll
    for (int j = 0; j < 4; ++j) {
      a0[j] = wmma_f16(a, frag_b_f16(B0t, D_, col0 + j * 16, k0), a0[j]);
      a1[j] = wmma_f16(a, frag_b_f16(B1t, D_, col0 + j * 16, k0), a1[j]);
    }
    a = an;
  }
  const int lane = threadIdx.x & 31;
  const int r = row0 + ((lane >> 4) << 3);
#pragma unroll
  for (int j = 0; j < 4; ++j) {
    const int c = col0 + j * 16 + (lane & 15);
    const float b0 = bi0[c], b1 = bi1[c];
#pragma unroll
    for (int i = 0; i < 8; ++i) {
      const float x0 = a0[j][i] + b0;
      const float x1 = a1[j][i] + b1;
      const float g = 0.5f * x0 * (1.f + erff(x0 * 0.70710678118654752f));
      HH[(size_t)(r + i) * FF_ + c] = (_Float16)(g * x1);
    }
  }
}

// ---------------------------------------------------------------------------
extern "C" void kernel_launch(void* const* d_in, const int* in_sizes, int n_in,
                              void* d_out, int out_size, void* d_ws, size_t ws_size,
                              hipStream_t stream) {
  const float* x    = (const float*)d_in[0];
  const float* wq   = (const float*)d_in[1];
  const float* bq   = (const float*)d_in[2];
  const float* wk   = (const float*)d_in[3];
  const float* bk   = (const float*)d_in[4];
  const float* wv   = (const float*)d_in[5];
  const float* bv   = (const float*)d_in[6];
  const float* wo   = (const float*)d_in[7];
  const float* bo   = (const float*)d_in[8];
  const float* ln1g = (const float*)d_in[9];
  const float* ln1b = (const float*)d_in[10];
  const float* wi0  = (const float*)d_in[11];
  const float* bi0  = (const float*)d_in[12];
  const float* wi1  = (const float*)d_in[13];
  const float* bi1  = (const float*)d_in[14];
  const float* wff  = (const float*)d_in[15];
  const float* bff  = (const float*)d_in[16];
  const float* ln2g = (const float*)d_in[17];
  const float* ln2b = (const float*)d_in[18];

  float* out  = (float*)d_out;                       // [B,S,D]
  float* attn = out + (size_t)B_ * S_ * D_;          // [B,H,S,S]

  char* ws = (char*)d_ws;
  const size_t MB = 1024 * 1024;
  _Float16* wqT   = (_Float16*)(ws + 0 * MB);        // [1024,1024]  2MB
  _Float16* wkT   = (_Float16*)(ws + 2 * MB);
  _Float16* wvT   = (_Float16*)(ws + 4 * MB);
  _Float16* woT   = (_Float16*)(ws + 6 * MB);
  _Float16* wi0T  = (_Float16*)(ws + 8 * MB);        // [4096,1024]  8MB
  _Float16* wi1T  = (_Float16*)(ws + 16 * MB);
  _Float16* wffT  = (_Float16*)(ws + 24 * MB);       // [1024,4096]  8MB
  _Float16* h16   = (_Float16*)(ws + 32 * MB);       // [4096,1024]  8MB
  _Float16* q16   = (_Float16*)(ws + 40 * MB);
  _Float16* k16   = (_Float16*)(ws + 48 * MB);
  _Float16* vT16  = (_Float16*)(ws + 56 * MB);       // [B,H,DK,S]
  _Float16* ctx16 = (_Float16*)(ws + 32 * MB);       // reuse h16
  float*    h1    = (float*)   (ws + 40 * MB);       // reuse q16+k16 (16MB)
  _Float16* ln216 = (_Float16*)(ws + 56 * MB);       // reuse vT16
  _Float16* hh16  = (_Float16*)(ws + 64 * MB);       // [4096,4096] 32MB

  // 1) weight convert + transpose to f16
  k_cvtw<<<(D_ * D_ + 255) / 256,  256, 0, stream>>>(wq,  wqT,  D_,  D_);
  k_cvtw<<<(D_ * D_ + 255) / 256,  256, 0, stream>>>(wk,  wkT,  D_,  D_);
  k_cvtw<<<(D_ * D_ + 255) / 256,  256, 0, stream>>>(wv,  wvT,  D_,  D_);
  k_cvtw<<<(D_ * D_ + 255) / 256,  256, 0, stream>>>(wo,  woT,  D_,  D_);
  k_cvtw<<<(D_ * FF_ + 255) / 256, 256, 0, stream>>>(wi0, wi0T, D_,  FF_);
  k_cvtw<<<(D_ * FF_ + 255) / 256, 256, 0, stream>>>(wi1, wi1T, D_,  FF_);
  k_cvtw<<<(FF_ * D_ + 255) / 256, 256, 0, stream>>>(wff, wffT, FF_, D_);

  // 2) LN1
  k_ln<<<NTOK, 256, 0, stream>>>(x, ln1g, ln1b, h16);

  // 3) Q,K,V projections (waves = 256*16 = 4096 -> 512 blocks)
  k_gemm_f16<<<512, 256, 0, stream>>>(h16, D_, wqT, D_, bq, q16, D_, 16, D_);
  k_gemm_f16<<<512, 256, 0, stream>>>(h16, D_, wkT, D_, bk, k16, D_, 16, D_);
  k_gemm_vT <<<512, 256, 0, stream>>>(h16, wvT, bv, vT16);

  // 4) scores -> attn region of d_out (128*32 waves per (b,h))
  k_scores<<<dim3(512, B_ * H_), 256, 0, stream>>>(q16, k16, attn);

  // 5) softmax in place (B*H*S rows)
  k_softmax<<<B_ * H_ * S_, 256, 0, stream>>>(attn);

  // 6) ctx = attn . V  (128 waves per (b,h) -> 16 blocks)
  k_ctx<<<dim3(16, B_ * H_), 256, 0, stream>>>(attn, vT16, ctx16);

  // 7) h1 = inputs + ctx @ wo + bo
  k_gemm_resid<<<512, 256, 0, stream>>>(ctx16, D_, woT, D_, bo, x, h1, D_, 16, D_);

  // 8) LN2
  k_ln<<<NTOK, 256, 0, stream>>>(h1, ln2g, ln2b, ln216);

  // 9) fused GeGLU (256*64 waves -> 2048 blocks)
  k_geglu<<<2048, 256, 0, stream>>>(ln216, wi0T, wi1T, bi0, bi1, hh16);

  // 10) out = h1 + hh @ wff + bff
  k_gemm_resid<<<512, 256, 0, stream>>>(hh16, FF_, wffT, FF_, bff, h1, out, D_, 16, FF_);
}
